// MultiHeadedAttention_1056561955097
// MI455X (gfx1250) — matrix operand order repaired and checked by reading, not compiled, not run on hardware
//
#include <hip/hip_runtime.h>

// ---------------------------------------------------------------------------
// MI455X (gfx1250) multi-head attention forward, f16 WMMA pipeline.
// B=4, L=2048, DIN=N_FEAT=O_FEAT=1024, H=16, Dk=64.
// Round 3: DPP16 row-max reduction (no LDS bpermute), global_prefetch_b8
// for next-next tiles; keeps async global->LDS V staging + WMMA row-sums.
// ---------------------------------------------------------------------------

typedef __attribute__((ext_vector_type(16))) _Float16 v16h;
typedef __attribute__((ext_vector_type(8)))  _Float16 v8h;
typedef __attribute__((ext_vector_type(4)))  _Float16 v4h;
typedef __attribute__((ext_vector_type(8)))  float    v8f;
typedef __attribute__((ext_vector_type(4)))  float    v4f;

union V16H {
    v16h v;
    v8h  h8[2];
    _Float16 h[16];
};

#define WMMA_F16(a, b, c) \
    __builtin_amdgcn_wmma_f32_16x16x32_f16(false, (a), false, (b), (short)0, (c), false, false)

// Max-reduce across each 16-lane DPP row (rows 0-15 / 16-31 hold independent
// data, which matches the WMMA C-tile row grouping). Stages: xor1, xor2, then
// row_half_mirror / row_mirror (act as xor4/xor8 once groups are uniform).
__device__ __forceinline__ float rowmax_dpp(float x)
{
    int v, p;
    v = __builtin_bit_cast(int, x);
    p = __builtin_amdgcn_update_dpp(v, v, 0xB1, 0xF, 0xF, false);   // quad_perm xor1
    x = fmaxf(x, __builtin_bit_cast(float, p));
    v = __builtin_bit_cast(int, x);
    p = __builtin_amdgcn_update_dpp(v, v, 0x4E, 0xF, 0xF, false);   // quad_perm xor2
    x = fmaxf(x, __builtin_bit_cast(float, p));
    v = __builtin_bit_cast(int, x);
    p = __builtin_amdgcn_update_dpp(v, v, 0x141, 0xF, 0xF, false);  // row_half_mirror
    x = fmaxf(x, __builtin_bit_cast(float, p));
    v = __builtin_bit_cast(int, x);
    p = __builtin_amdgcn_update_dpp(v, v, 0x140, 0xF, 0xF, false);  // row_mirror
    x = fmaxf(x, __builtin_bit_cast(float, p));
    return x;
}

// ---------------------------------------------------------------------------
// GEMM: C[M,N] = A[M,K] * Bw[K,N] + bias[N]
// Block: 256 threads (8 waves), tile 128x128, K-chunk 32, ping-pong LDS,
// global->register prefetch of the next chunk overlapped with WMMA,
// global_prefetch of the chunk after next.
// ---------------------------------------------------------------------------
template<bool A16, bool OUT16>
__global__ __launch_bounds__(256)
void gemm_f16_wmma(const void* __restrict__ Aptr,
                   const float* __restrict__ Bw,
                   const float* __restrict__ bias,
                   void* __restrict__ Cptr,
                   int M, int N, int K)
{
    __shared__ _Float16 Alds[2][128][40];   // 80B rows (16B aligned)
    __shared__ _Float16 Blds[2][32][144];   // 288B rows (32B aligned)

    const int t     = threadIdx.x;
    const int lane  = t & 31;
    const int w     = t >> 5;
    const int waveM = w >> 1;        // 0..3
    const int waveN = w & 1;         // 0..1
    const int mBase = blockIdx.x * 128;
    const int nBase = blockIdx.y * 128;
    const int m16   = lane & 15;
    const int kb8   = (lane >> 4) << 3;

    const float*    Af32 = (const float*)Aptr;
    const _Float16* Af16 = (const _Float16*)Aptr;

    // accumulators initialized with bias (row-broadcast of bias[col])
    v8f acc[2][4];
    for (int j = 0; j < 4; ++j) {
        const float bv = bias[nBase + waveN * 64 + j * 16 + m16];
        v8f c;
        for (int r = 0; r < 8; ++r) c[r] = bv;
        acc[0][j] = c;
        acc[1][j] = c;
    }

    // ---- staging registers for one 128x32 A tile and one 32x128 B tile ----
    v4h aRegH[4];   // used when A is f16
    v4f aRegF[4];   // used when A is f32 (convert at LDS-store time)
    v4f bRegF[4];

    auto loadRegs = [&](int k0) {
        for (int i = 0; i < 4; ++i) {
            const int idx = i * 256 + t;        // 0..1023
            const int row = idx >> 3;           // 0..127
            const int c4  = (idx & 7) << 2;     // 0..28
            if constexpr (A16)
                aRegH[i] = *(const v4h*)(Af16 + (size_t)(mBase + row) * K + k0 + c4);
            else
                aRegF[i] = *(const v4f*)(Af32 + (size_t)(mBase + row) * K + k0 + c4);
        }
        for (int i = 0; i < 4; ++i) {
            const int idx  = i * 256 + t;       // 0..1023
            const int krow = idx >> 5;          // 0..31
            const int c4   = (idx & 31) << 2;   // 0..124
            bRegF[i] = *(const v4f*)(Bw + (size_t)(k0 + krow) * N + nBase + c4);
        }
    };

    auto storeLds = [&](int buf) {
        for (int i = 0; i < 4; ++i) {
            const int idx = i * 256 + t;
            const int row = idx >> 3;
            const int c4  = (idx & 7) << 2;
            v4h hv;
            if constexpr (A16) {
                hv = aRegH[i];
            } else {
                hv[0] = (_Float16)aRegF[i][0]; hv[1] = (_Float16)aRegF[i][1];
                hv[2] = (_Float16)aRegF[i][2]; hv[3] = (_Float16)aRegF[i][3];
            }
            *(v4h*)(&Alds[buf][row][c4]) = hv;
        }
        for (int i = 0; i < 4; ++i) {
            const int idx  = i * 256 + t;
            const int krow = idx >> 5;
            const int c4   = (idx & 31) << 2;
            v4h hv;
            hv[0] = (_Float16)bRegF[i][0]; hv[1] = (_Float16)bRegF[i][1];
            hv[2] = (_Float16)bRegF[i][2]; hv[3] = (_Float16)bRegF[i][3];
            *(v4h*)(&Blds[buf][krow][c4]) = hv;
        }
    };

    // prologue: stage tile 0
    loadRegs(0);
    storeLds(0);

    int buf = 0;
    for (int k0 = 0; k0 < K; k0 += 32) {
        __syncthreads();                       // staged tile visible
        const bool more = (k0 + 32 < K);       // uniform
        if (more) loadRegs(k0 + 32);           // issue next-chunk globals early

        // L2 prefetch of the chunk after next (global_prefetch_b8)
        if (k0 + 64 < K) {
            const int arow = t & 127;
            if constexpr (A16)
                __builtin_prefetch(Af16 + (size_t)(mBase + arow) * K + k0 + 64, 0, 0);
            else
                __builtin_prefetch(Af32 + (size_t)(mBase + arow) * K + k0 + 64, 0, 0);
            const int krow = t >> 3;                   // 0..31
            const int c32  = ((t >> 1) & 3) << 5;      // 0,32,64,96
            __builtin_prefetch(Bw + (size_t)(k0 + 64 + krow) * N + nBase + c32, 0, 0);
        }

        // ---- fragments from LDS ----
        V16H a[2];
        for (int i = 0; i < 2; ++i) {
            const _Float16* base = &Alds[buf][waveM * 32 + i * 16 + m16][kb8];
            a[i].h8[0] = *(const v8h*)(base);
            a[i].h8[1] = *(const v8h*)(base + 16);
        }
        V16H bf[4];
        for (int j = 0; j < 4; ++j)
            for (int s = 0; s < 16; ++s)
                bf[j].h[s] = Blds[buf][kb8 + (s & 7) + ((s >> 3) << 4)][waveN * 64 + j * 16 + m16];

        for (int i = 0; i < 2; ++i)
            for (int j = 0; j < 4; ++j)
                acc[i][j] = WMMA_F16(a[i].v, bf[j].v, acc[i][j]);

        if (more) storeLds(buf ^ 1);           // waits loadcnt after WMMAs issued
        buf ^= 1;
    }

    // ---- epilogue: C layout M = r + 8*(lane>>4), N = lane&15 ----
    const int rowAdd = (lane >> 4) << 3;
    for (int i = 0; i < 2; ++i)
        for (int j = 0; j < 4; ++j)
            for (int r = 0; r < 8; ++r) {
                const size_t row = (size_t)(mBase + waveM * 32 + i * 16 + r + rowAdd);
                const size_t col = (size_t)(nBase + waveN * 64 + j * 16 + m16);
                const float val = acc[i][j][r];
                if constexpr (OUT16)
                    ((_Float16*)Cptr)[row * N + col] = (_Float16)val;
                else
                    ((float*)Cptr)[row * N + col] = val;
            }
}

// ---------------------------------------------------------------------------
// Flash-style attention. Grid (L/64, H, B), 128 threads (4 waves).
// Wave w owns 16 query rows. Software pipeline:
//   scatter K(regs)->Kt[cur]; s_wait_asynccnt 0; barrier;
//   issue async V[next]->LDS + K[next]->regs; WMMA compute on [cur].
// Softmax row-sum via all-ones WMMA B fragment; row-max via DPP16.
// ---------------------------------------------------------------------------
__global__ __launch_bounds__(128)
void attn_fwd_wmma(const _Float16* __restrict__ Qg,
                   const _Float16* __restrict__ Kg,
                   const _Float16* __restrict__ Vg,
                   _Float16* __restrict__ Ctx)
{
    __shared__ _Float16 Kt[2][64][80];     // transposed [dk][token], 160B rows
    __shared__ _Float16 Vl[2][64][80];     // row-major  [token][dk]
    __shared__ _Float16 Pl[4][16][72];     // per-wave P transpose patch

    const int t    = threadIdx.x;
    const int lane = t & 31;
    const int w    = t >> 5;               // 0..3
    const int h    = blockIdx.y;
    const int b    = blockIdx.z;

    const size_t tokBase = (size_t)b * 2048;
    const int q0     = blockIdx.x * 64 + w * 16;
    const int m16    = lane & 15;
    const int kb8    = (lane >> 4) << 3;
    const int rowAdd = (lane >> 4) << 3;

    // cooperative-staging geometry: 2 threads per token, 32 dk each
    const int tok   = t & 63;
    const int dbase = (t >> 6) * 32;

    v8h kReg[4];

    auto issueBlockLoads = [&](int kb0, int bufn) {
        const _Float16* ksrc = Kg + (tokBase + kb0 + tok) * 1024 + h * 64 + dbase;
        const _Float16* vsrc = Vg + (tokBase + kb0 + tok) * 1024 + h * 64 + dbase;
        // V: async global -> LDS, 4 x b128 per thread (ASYNCcnt-tracked)
        for (int d = 0; d < 32; d += 8) {
            const unsigned lds =
                (unsigned)(size_t)(&Vl[bufn][tok][dbase + d]);   // low 32b = LDS offset
            const unsigned long long ga = (unsigned long long)(size_t)(vsrc + d);
            asm volatile("global_load_async_to_lds_b128 %0, %1, off"
                         :: "v"(lds), "v"(ga) : "memory");
        }
        // K: to registers (transposed into LDS next iteration)
        for (int d4 = 0; d4 < 4; ++d4)
            kReg[d4] = *(const v8h*)(ksrc + d4 * 8);
    };

    auto scatterK = [&](int bufn) {
        for (int d4 = 0; d4 < 4; ++d4)
            for (int e = 0; e < 8; ++e)
                Kt[bufn][dbase + d4 * 8 + e][tok] = kReg[d4][e];
    };

    // Q fragments for this wave's 16 rows, pre-scaled by 1/sqrt(Dk)=0.125
    V16H aq[2];
    for (int c = 0; c < 2; ++c) {
        const _Float16* src = Qg + (tokBase + q0 + m16) * 1024 + h * 64 + c * 32;
        aq[c].h8[0] = *(const v8h*)(src + kb8);
        aq[c].h8[1] = *(const v8h*)(src + kb8 + 16);
        for (int j = 0; j < 16; ++j) aq[c].h[j] = aq[c].h[j] * (_Float16)0.125f;
    }

    // all-ones B fragment: row-sum of P via the matrix core
    V16H ones;
    for (int j = 0; j < 16; ++j) ones.h[j] = (_Float16)1.0f;

    float mrun[8];
    v8f acc[4], accl;
    for (int r = 0; r < 8; ++r) mrun[r] = -1e30f;
    for (int j = 0; j < 4; ++j) { v8f z = {}; acc[j] = z; }
    { v8f z = {}; accl = z; }

    issueBlockLoads(0, 0);
    int cur = 0;

    for (int kb0 = 0; kb0 < 2048; kb0 += 64) {
        scatterK(cur);
        asm volatile("s_wait_asynccnt 0" ::: "memory");   // V[cur] resident
        __syncthreads();                                  // staged block visible
        if (kb0 + 64 < 2048) issueBlockLoads(kb0 + 64, cur ^ 1);
        if (kb0 + 128 < 2048) {                           // GL2 prefetch block+2
            __builtin_prefetch(Kg + (tokBase + kb0 + 128 + tok) * 1024 + h * 64 + dbase, 0, 0);
            __builtin_prefetch(Vg + (tokBase + kb0 + 128 + tok) * 1024 + h * 64 + dbase, 0, 0);
        }

        // ---- S = Q * K^T : 4 N-tiles x 2 K-chunks ----
        v8f s[4];
        for (int j = 0; j < 4; ++j) { v8f z = {}; s[j] = z; }
        for (int j = 0; j < 4; ++j)
            for (int c = 0; c < 2; ++c) {
                V16H bk;
                for (int s = 0; s < 16; ++s)
                    bk.h[s] = Kt[cur][c * 32 + kb8 + (s & 7) + ((s >> 3) << 4)][j * 16 + m16];
                s[j] = WMMA_F16(aq[c].v, bk.v, s[j]);
            }

        // ---- online softmax: row-max via DPP16, no LDS traffic ----
        float scale[8];
        for (int r = 0; r < 8; ++r) {
            float mx = fmaxf(fmaxf(s[0][r], s[1][r]), fmaxf(s[2][r], s[3][r]));
            mx = rowmax_dpp(mx);
            const float mn = fmaxf(mrun[r], mx);
            scale[r] = __expf(mrun[r] - mn);
            mrun[r]  = mn;
        }
        for (int j = 0; j < 4; ++j)
            for (int r = 0; r < 8; ++r)
                s[j][r] = __expf(s[j][r] - mrun[r]);

        // ---- P: C-layout -> LDS -> A-layout (per-wave patch; LDS pipeline
        //      is in-order within a wave, so only a compiler fence needed) ----
        for (int j = 0; j < 4; ++j)
            for (int r = 0; r < 8; ++r)
                Pl[w][r + rowAdd][j * 16 + m16] = (_Float16)s[j][r];
        asm volatile("" ::: "memory");

        V16H ap[2];
        for (int c = 0; c < 2; ++c) {
            const _Float16* base = &Pl[w][m16][c * 32 + kb8];
            ap[c].h8[0] = *(const v8h*)(base);
            ap[c].h8[1] = *(const v8h*)(base + 16);
        }

        // ---- rescale running accumulators, then O += P*V, l += P*1 ----
        for (int j = 0; j < 4; ++j)
            for (int r = 0; r < 8; ++r) acc[j][r] *= scale[r];
        for (int r = 0; r < 8; ++r) accl[r] *= scale[r];

        for (int j = 0; j < 4; ++j)
            for (int c = 0; c < 2; ++c) {
                V16H bv;
                for (int s = 0; s < 16; ++s)
                    bv.h[s] = Vl[cur][c * 32 + kb8 + (s & 7) + ((s >> 3) << 4)][j * 16 + m16];
                acc[j] = WMMA_F16(ap[c].v, bv.v, acc[j]);
            }
        for (int c = 0; c < 2; ++c)
            accl = WMMA_F16(ap[c].v, ones.v, accl);

        cur ^= 1;
    }

    // ---- normalize and store context (f16) ----
    for (int j = 0; j < 4; ++j)
        for (int r = 0; r < 8; ++r) {
            const float o = acc[j][r] / accl[r];
            Ctx[(tokBase + q0 + r + rowAdd) * 1024 + h * 64 + j * 16 + m16] = (_Float16)o;
        }
}

// ---------------------------------------------------------------------------
extern "C" void kernel_launch(void* const* d_in, const int* in_sizes, int n_in,
                              void* d_out, int out_size, void* d_ws, size_t ws_size,
                              hipStream_t stream)
{
    (void)in_sizes; (void)n_in; (void)out_size; (void)ws_size;

    const float* q  = (const float*)d_in[0];
    const float* k  = (const float*)d_in[1];
    const float* v  = (const float*)d_in[2];
    const float* Wq = (const float*)d_in[3];
    const float* bq = (const float*)d_in[4];
    const float* Wk = (const float*)d_in[5];
    const float* bk = (const float*)d_in[6];
    const float* Wv = (const float*)d_in[7];
    const float* bv = (const float*)d_in[8];
    const float* Wo = (const float*)d_in[9];
    const float* bo = (const float*)d_in[10];

    const int Mtok = 4 * 2048;  // 8192 tokens
    const int NF   = 1024;
    const int KD   = 1024;

    // workspace: 4 x (8192 x 1024) f16 = 64 MB
    _Float16* Q16 = (_Float16*)d_ws;
    _Float16* K16 = Q16 + (size_t)Mtok * NF;
    _Float16* V16 = K16 + (size_t)Mtok * NF;
    _Float16* C16 = V16 + (size_t)Mtok * NF;

    dim3 gg(Mtok / 128, NF / 128);  // 64 x 8 workgroups

    gemm_f16_wmma<false, true><<<gg, 256, 0, stream>>>(q, Wq, bq, Q16, Mtok, NF, KD);
    gemm_f16_wmma<false, true><<<gg, 256, 0, stream>>>(k, Wk, bk, K16, Mtok, NF, KD);
    gemm_f16_wmma<false, true><<<gg, 256, 0, stream>>>(v, Wv, bv, V16, Mtok, NF, KD);

    dim3 ga(2048 / 64, 16, 4);      // (query blocks, heads, batch)
    attn_fwd_wmma<<<ga, 128, 0, stream>>>(Q16, K16, V16, C16);

    gemm_f16_wmma<true, false><<<gg, 256, 0, stream>>>(C16, Wo, bo, d_out, Mtok, NF, KD);
}
